// VectorQuantizer_90881507984102
// MI455X (gfx1250) — matrix operand order repaired and checked by reading, not compiled, not run on hardware
//
#include <hip/hip_runtime.h>

// ---------------------------------------------------------------------------
// VQ-VAE quantizer for MI455X (gfx1250, wave32, WMMA).
// Score GEMM runs as bf16x3-split V_WMMA_F32_16X16X32_BF16 (f32-class accuracy
// at 8/3x the f32-WMMA rate). Argmin only needs ||e||^2 - 2 x.e (row-constant
// ||x||^2 dropped).
// ---------------------------------------------------------------------------

typedef __attribute__((ext_vector_type(16))) __bf16 v16bf;
typedef __attribute__((ext_vector_type(8)))  float  v8f;
typedef __attribute__((ext_vector_type(4)))  float  v4f;

#define DDIM    256
#define KCODES  8192
#define NROWS   32768
#define NTILES  (KCODES / 16)   // 512 column tiles of 16 codes

// ---------------------------------------------------------------------------
// K0: split embeddings into bf16 hi/lo tables + row norms ||e_k||^2
// grid = KCODES blocks x 256 threads (one element per thread)
// ---------------------------------------------------------------------------
__global__ __launch_bounds__(256) void vq_prep(const float* __restrict__ E,
                                               __bf16* __restrict__ Eh,
                                               __bf16* __restrict__ El,
                                               float* __restrict__ enorm) {
    const int row = blockIdx.x;
    const int t   = threadIdx.x;
    const float e = E[(size_t)row * DDIM + t];
    const __bf16 hi = (__bf16)e;
    const __bf16 lo = (__bf16)(e - (float)hi);
    Eh[(size_t)row * DDIM + t] = hi;
    El[(size_t)row * DDIM + t] = lo;

    float s = e * e;
#pragma unroll
    for (int off = 16; off > 0; off >>= 1) s += __shfl_xor(s, off, 32);
    __shared__ float sm[8];
    if ((t & 31) == 0) sm[t >> 5] = s;
    __syncthreads();
    if (t == 0) {
        float tot = 0.f;
#pragma unroll
        for (int i = 0; i < 8; ++i) tot += sm[i];
        enorm[row] = tot;
    }
}

// ---------------------------------------------------------------------------
// zero the histogram + sse accumulator (graph-safe: no memset)
// ---------------------------------------------------------------------------
__global__ void vq_zero(unsigned* __restrict__ hist, float* __restrict__ sse) {
    const int i = blockIdx.x * blockDim.x + threadIdx.x;
    if (i < KCODES) hist[i] = 0u;
    if (i == 0) *sse = 0.f;
}

// ---------------------------------------------------------------------------
// K1: WMMA argmin kernel.
// Block = 256 threads = 8 waves; each wave owns 16 input rows.
// A fragments (x hi/lo, 8 K-blocks) are built once and live in VGPRs for the
// whole 512-tile sweep over the codebook. B fragments load 32 contiguous
// bytes/lane from the L2-resident bf16 tables (contiguous k order used for A and B here:
// lane = N + 16*(K/16); element e -> K = 16*(lane/16) + e).
// C layout: lanes 0-15 -> M=r, lanes 16-31 -> M=r+8; N = lane & 15.
// ---------------------------------------------------------------------------
__global__ __launch_bounds__(256, 1) void vq_argmin(
        const float*   __restrict__ X,
        const __bf16*  __restrict__ Eh,
        const __bf16*  __restrict__ El,
        const float*   __restrict__ enorm,
        unsigned*      __restrict__ idx_out) {
    const int lane = threadIdx.x & 31;
    const int wave = threadIdx.x >> 5;
    const int m16  = lane & 15;     // A: M row; B: N col; C: N col
    const int h    = lane >> 4;     // A: K-half; B: K-group; C: M+8 select
    const int row  = blockIdx.x * 128 + wave * 16 + m16;

    // ---- build A fragments (hi/lo split) in registers ----
    v16bf xh[8], xl[8];
    const float* xr = X + (size_t)row * DDIM;
#pragma unroll
    for (int kb = 0; kb < 8; ++kb) {
        const float* p = xr + kb * 32 + h * 16;
        v4f q0 = *(const v4f*)(p);
        v4f q1 = *(const v4f*)(p + 4);
        v4f q2 = *(const v4f*)(p + 8);
        v4f q3 = *(const v4f*)(p + 12);
        float a[16] = {q0.x, q0.y, q0.z, q0.w, q1.x, q1.y, q1.z, q1.w,
                       q2.x, q2.y, q2.z, q2.w, q3.x, q3.y, q3.z, q3.w};
#pragma unroll
        for (int e = 0; e < 16; ++e) {
            __bf16 hi = (__bf16)a[e];
            xh[kb][e] = hi;
            xl[kb][e] = (__bf16)(a[e] - (float)hi);
        }
    }

    float    minv[8];
    unsigned mini[8];
#pragma unroll
    for (int r = 0; r < 8; ++r) { minv[r] = 3.4e38f; mini[r] = 0u; }

    // ---- sweep the codebook: 512 tiles of 16 codes ----
    for (int t = 0; t < NTILES; ++t) {
        const float en = enorm[t * 16 + m16];
        const size_t boff = (size_t)(t * 16 + m16) * DDIM + h * 16;
        const __bf16* eh = Eh + boff;
        const __bf16* el = El + boff;

        v8f acc = {0.f, 0.f, 0.f, 0.f, 0.f, 0.f, 0.f, 0.f};
#pragma unroll
        for (int kb = 0; kb < 8; ++kb) {
            v16bf bh = *(const v16bf*)(eh + kb * 32);
            v16bf bl = *(const v16bf*)(el + kb * 32);
            acc = __builtin_amdgcn_wmma_f32_16x16x32_bf16(
                false, xh[kb], false, bh, (short)0, acc, false, false);
            acc = __builtin_amdgcn_wmma_f32_16x16x32_bf16(
                false, xh[kb], false, bl, (short)0, acc, false, false);
            acc = __builtin_amdgcn_wmma_f32_16x16x32_bf16(
                false, xl[kb], false, bh, (short)0, acc, false, false);
        }

        const unsigned col = (unsigned)(t * 16 + m16);
#pragma unroll
        for (int r = 0; r < 8; ++r) {
            float dist = __builtin_fmaf(-2.0f, acc[r], en);
            bool lt = dist < minv[r];           // strict <: keeps first index
            minv[r] = lt ? dist : minv[r];
            mini[r] = lt ? col  : mini[r];
        }
    }

    // ---- reduce (min, argmin) across the 16 lanes of each half ----
#pragma unroll
    for (int r = 0; r < 8; ++r) {
        float    v = minv[r];
        unsigned i = mini[r];
#pragma unroll
        for (int off = 1; off < 16; off <<= 1) {
            float    ov = __shfl_xor(v, off, 32);
            unsigned oi = (unsigned)__shfl_xor((int)i, off, 32);
            if (ov < v || (ov == v && oi < i)) { v = ov; i = oi; }
        }
        if (m16 == 0) {
            const int orow = blockIdx.x * 128 + wave * 16 + r + (h << 3);
            idx_out[orow] = i;
        }
    }
}

// ---------------------------------------------------------------------------
// K2: gather quantized rows (straight-through value == quantized), SSE,
// histogram, float-cast indices. One row per block.
// ---------------------------------------------------------------------------
__global__ __launch_bounds__(256) void vq_gather(
        const float* __restrict__ X, const float* __restrict__ E,
        const unsigned* __restrict__ idx,
        float* __restrict__ qout, float* __restrict__ idxf,
        float* __restrict__ sse, unsigned* __restrict__ hist) {
    const int row = blockIdx.x;
    const int d   = threadIdx.x;
    const unsigned k = idx[row];
    const float x = X[(size_t)row * DDIM + d];
    const float q = E[(size_t)k * DDIM + d];
    qout[(size_t)row * DDIM + d] = q;
    const float diff = x - q;
    float s = diff * diff;
#pragma unroll
    for (int off = 16; off > 0; off >>= 1) s += __shfl_xor(s, off, 32);
    __shared__ float sm[8];
    if ((d & 31) == 0) sm[d >> 5] = s;
    __syncthreads();
    if (d == 0) {
        float tot = 0.f;
#pragma unroll
        for (int i = 0; i < 8; ++i) tot += sm[i];
        atomicAdd(sse, tot);
        atomicAdd(&hist[k], 1u);
        idxf[row] = (float)k;
    }
}

// ---------------------------------------------------------------------------
// K3: vq_loss = 1.25 * MSE ; perplexity = exp(-sum p log(p + 1e-10))
// ---------------------------------------------------------------------------
__global__ __launch_bounds__(256) void vq_final(
        const unsigned* __restrict__ hist, const float* __restrict__ sse,
        float* __restrict__ loss_out, float* __restrict__ perp_out) {
    const int t = threadIdx.x;
    float s = 0.f;
    for (int b = t; b < KCODES; b += 256) {
        float p = (float)hist[b] * (1.0f / (float)NROWS);
        s += p * __logf(p + 1e-10f);
    }
#pragma unroll
    for (int off = 16; off > 0; off >>= 1) s += __shfl_xor(s, off, 32);
    __shared__ float sm[8];
    if ((t & 31) == 0) sm[t >> 5] = s;
    __syncthreads();
    if (t == 0) {
        float tot = 0.f;
#pragma unroll
        for (int i = 0; i < 8; ++i) tot += sm[i];
        *perp_out = __expf(-tot);
        *loss_out = 1.25f * (*sse) * (1.0f / ((float)NROWS * (float)DDIM));
    }
}

// ---------------------------------------------------------------------------
// launch: d_in[0] = inputs f32 [32,32,32,256], d_in[1] = embeddings f32 [8192,256]
// d_out = [quantized 8388608][vq_loss 1][perplexity 1][indices 32768] f32
// ---------------------------------------------------------------------------
extern "C" void kernel_launch(void* const* d_in, const int* in_sizes, int n_in,
                              void* d_out, int out_size, void* d_ws, size_t ws_size,
                              hipStream_t stream) {
    const float* X = (const float*)d_in[0];
    const float* E = (const float*)d_in[1];

    float* out  = (float*)d_out;
    float* qout = out;                      // 8388608
    float* loss = out + (size_t)NROWS * DDIM;
    float* perp = loss + 1;
    float* idxf = perp + 1;                 // 32768

    char* ws = (char*)d_ws;
    __bf16*   Eh    = (__bf16*)(ws);                        // 4 MiB
    __bf16*   El    = (__bf16*)(ws + (size_t)4194304);      // 4 MiB
    float*    enorm = (float*) (ws + (size_t)8388608);      // 32 KiB
    unsigned* idx   = (unsigned*)(ws + (size_t)8421376);    // 128 KiB
    unsigned* hist  = (unsigned*)(ws + (size_t)8552448);    // 32 KiB
    float*    sse   = (float*) (ws + (size_t)8585216);      // 4 B

    vq_prep  <<<KCODES,      256, 0, stream>>>(E, Eh, El, enorm);
    vq_zero  <<<KCODES / 256, 256, 0, stream>>>(hist, sse);
    vq_argmin<<<NROWS / 128, 256, 0, stream>>>(X, Eh, El, enorm, idx);
    vq_gather<<<NROWS,       256, 0, stream>>>(X, E, idx, qout, idxf, sse, hist);
    vq_final <<<1,           256, 0, stream>>>(hist, sse, loss, perp);
}
